// LSTMDecode_19353122636499
// MI455X (gfx1250) — matrix operand order repaired and checked by reading, not compiled, not run on hardware
//
#include <hip/hip_runtime.h>

// ---------------- types ----------------
typedef __attribute__((ext_vector_type(16))) __bf16 v16bf;
typedef __attribute__((ext_vector_type(8)))  float  v8f;
typedef unsigned short u16t;
typedef unsigned int   u32t;

union Frag { v16bf v; uint4 q[2]; };

__device__ __forceinline__ v8f wmma_bf16(v16bf a, v16bf b, v8f c) {
    return __builtin_amdgcn_wmma_f32_16x16x32_bf16(false, a, false, b, (short)0, c, false, false);
}

__device__ __forceinline__ u16t f2bf(float x) {
    u32t u = __float_as_uint(x);
    u32t r = ((u >> 16) & 1u) + 0x7FFFu;     // round to nearest even
    return (u16t)((u + r) >> 16);
}
__device__ __forceinline__ u32t pack2(float a, float b) {
    return (u32t)f2bf(a) | ((u32t)f2bf(b) << 16);
}
__device__ __forceinline__ float fsig(float x) { return 1.0f / (1.0f + __expf(-x)); }
__device__ __forceinline__ float ftanh(float x) {
    float e = __expf(2.0f * x);
    return 1.0f - 2.0f / (e + 1.0f);
}

// ---------------- workspace layout (bytes) ----------------
static const size_t OFF_W0  = 0;           // 24*64 frags * 1KB = 1,572,864
static const size_t OFF_W1  = 1572864;     // 16*64 * 1KB = 1,048,576
static const size_t OFF_WH  = 2621440;     // 8*8   * 1KB = 65,536
static const size_t OFF_B0  = 2686976;     // 1024 f32
static const size_t OFF_B1  = 2691072;     // 1024 f32
static const size_t OFF_SUM = 2695168;     // 128 f32
static const size_t OFF_SQ  = 2695680;     // 128 f32
static const size_t OFF_BNA = 2696192;     // 128 f32
static const size_t OFF_BNB = 2696704;     // 128 f32
static const size_t OFF_OUTA= 4194304;     // 8192 M-tiles * 8KB = 64MB (packed A frags of h2, t-major)
static const size_t OFF_Y   = 71303168;    // 131072*128 f32 = 64MB

// LDS pitches (elements)
#define P0 776     // 768 + 8 pad  (bf16)  -> 1552B row stride (conflict-free A loads)
#define P1 520     // 512 + 8 pad  (bf16)
#define ZP 1032    // 1024 + 8 pad (f32)

// ---------------- weight packing + bias fusion ----------------
__global__ void pack_kernel(const float* __restrict__ Wih0, const float* __restrict__ Whh0,
                            const float* __restrict__ bih0, const float* __restrict__ bhh0,
                            const float* __restrict__ Wih1, const float* __restrict__ Whh1,
                            const float* __restrict__ bih1, const float* __restrict__ bhh1,
                            const float* __restrict__ W1,
                            u32t* __restrict__ w0, u32t* __restrict__ w1, u32t* __restrict__ wh,
                            float* __restrict__ bias0, float* __restrict__ bias1,
                            float* __restrict__ gsum, float* __restrict__ gsq)
{
    const int tid = blockIdx.x * blockDim.x + threadIdx.x;
    const int N0 = 24 * 64 * 256;   // layer0 dwords
    const int N1 = 16 * 64 * 256;   // layer1 dwords
    const int N2 = 8 * 8 * 256;     // head dwords
    if (tid < N0) {
        int d = tid;
        int kb = d / (64 * 256); int rem = d % (64 * 256);
        int nt = rem / 256;      int r2  = rem % 256;
        int ln = r2 >> 3;        int v   = r2 & 7;
        int k = kb * 32 + ((v < 4) ? 2 * v : 16 + 2 * (v - 4)) + ((ln < 16) ? 0 : 8);
        int n = nt * 16 + (ln & 15);
        float f0, f1;
        if (k < 512) { f0 = Wih0[n * 512 + k];        f1 = Wih0[n * 512 + k + 1]; }
        else         { f0 = Whh0[n * 256 + (k - 512)]; f1 = Whh0[n * 256 + (k - 511)]; }
        w0[tid] = pack2(f0, f1);
    } else if (tid < N0 + N1) {
        int d = tid - N0;
        int kb = d / (64 * 256); int rem = d % (64 * 256);
        int nt = rem / 256;      int r2  = rem % 256;
        int ln = r2 >> 3;        int v   = r2 & 7;
        int k = kb * 32 + ((v < 4) ? 2 * v : 16 + 2 * (v - 4)) + ((ln < 16) ? 0 : 8);
        int n = nt * 16 + (ln & 15);
        float f0, f1;
        if (k < 256) { f0 = Wih1[n * 256 + k];        f1 = Wih1[n * 256 + k + 1]; }
        else         { f0 = Whh1[n * 256 + (k - 256)]; f1 = Whh1[n * 256 + (k - 255)]; }
        w1[d] = pack2(f0, f1);
    } else if (tid < N0 + N1 + N2) {
        int d = tid - N0 - N1;
        int kb = d / (8 * 256); int rem = d % (8 * 256);
        int nt = rem / 256;     int r2  = rem % 256;
        int ln = r2 >> 3;       int v   = r2 & 7;
        int k = kb * 32 + ((v < 4) ? 2 * v : 16 + 2 * (v - 4)) + ((ln < 16) ? 0 : 8);
        int n = nt * 16 + (ln & 15);
        wh[d] = pack2(W1[n * 256 + k], W1[n * 256 + k + 1]);
    } else if (tid < N0 + N1 + N2 + 1024) {
        int i = tid - (N0 + N1 + N2);
        bias0[i] = bih0[i] + bhh0[i];
    } else if (tid < N0 + N1 + N2 + 2048) {
        int i = tid - (N0 + N1 + N2 + 1024);
        bias1[i] = bih1[i] + bhh1[i];
    } else if (tid < N0 + N1 + N2 + 2048 + 256) {
        int i = tid - (N0 + N1 + N2 + 2048);
        if (i < 128) gsum[i] = 0.0f; else gsq[i - 128] = 0.0f;
    }
}

// ---------------- per-wave GEMM: (16 x 32K) @ (32K x 64) into z-buffer ----------------
__device__ __forceinline__ void gemm_tile(const u16t* __restrict__ Abuf, int pitch, int KB,
                                          const uint4* __restrict__ wfrag,
                                          const float* __restrict__ bias,
                                          float* __restrict__ zbuf,
                                          int lane, int wave)
{
    v8f acc0 = {}; v8f acc1 = {}; v8f acc2 = {}; v8f acc3 = {};
    const int nt0 = wave * 4;
    const int row = lane & 15;
    const int koff = (lane < 16) ? 0 : 8;
    const u16t* abase = Abuf + row * pitch + koff;
    for (int kb = 0; kb < KB; ++kb) {
        Frag a;
        const u16t* ap = abase + kb * 32;
        a.q[0] = *(const uint4*)(ap);         // K = kb*32 + koff + 0..7
        a.q[1] = *(const uint4*)(ap + 16);    // K = kb*32 + koff + 16..23
        const uint4* wp = wfrag + (size_t)(kb * 64 + nt0) * 64 + lane * 2;
        Frag b0, b1, b2, b3;
        b0.q[0] = wp[0];   b0.q[1] = wp[1];
        b1.q[0] = wp[64];  b1.q[1] = wp[65];
        b2.q[0] = wp[128]; b2.q[1] = wp[129];
        b3.q[0] = wp[192]; b3.q[1] = wp[193];
        acc0 = wmma_bf16(a.v, b0.v, acc0);
        acc1 = wmma_bf16(a.v, b1.v, acc1);
        acc2 = wmma_bf16(a.v, b2.v, acc2);
        acc3 = wmma_bf16(a.v, b3.v, acc3);
    }
    const int mlo = (lane < 16) ? 0 : 8;
    const int nb = nt0 * 16 + (lane & 15);
    const float bb0 = bias[nb], bb1 = bias[nb + 16], bb2 = bias[nb + 32], bb3 = bias[nb + 48];
#pragma unroll
    for (int v = 0; v < 8; ++v) {
        int m = mlo + v;
        zbuf[m * ZP + nb]      = acc0[v] + bb0;
        zbuf[m * ZP + nb + 16] = acc1[v] + bb1;
        zbuf[m * ZP + nb + 32] = acc2[v] + bb2;
        zbuf[m * ZP + nb + 48] = acc3[v] + bb3;
    }
}

// ---------------- persistent LSTM decode: 16 WGs, one 16-row batch tile each ----------------
__global__ __launch_bounds__(512) void lstm_kernel(const float* __restrict__ audio,
                                                   const float* __restrict__ s_rt,
                                                   const uint4* __restrict__ w0,
                                                   const uint4* __restrict__ w1,
                                                   const float* __restrict__ bias0,
                                                   const float* __restrict__ bias1,
                                                   u32t* __restrict__ outA)
{
    extern __shared__ char smem[];
    u16t*  xbuf  = (u16t*)smem;                         // 16 x P0 bf16  [prev_out | audio | h1]
    u16t*  l1buf = xbuf + 16 * P0;                      // 16 x P1 bf16  [h1n | h2prev]
    float* zbuf  = (float*)(smem + 41472);              // 16 x ZP f32
    u16t*  h2st  = (u16t*)(smem + 41472 + 16 * ZP * 4); // 16 x 256 bf16

    const int tid  = threadIdx.x;
    const int lane = tid & 31;
    const int wave = tid >> 5;         // 0..15 ; also epilogue row
    const int wg   = blockIdx.x;       // 0..15
    const int b0   = wg * 16;
    const int rr   = wave;
    const int uu   = lane * 8;

    float c1[8], c2[8];
#pragma unroll
    for (int j = 0; j < 8; ++j) { c1[j] = 0.0f; c2[j] = 0.0f; }

    // init: prev_out = mean(s_rt) ; h1 = 0 ; h2prev = 0
#pragma unroll
    for (int j = 0; j < 8; ++j) {
        int u = uu + j;
        float s = 0.0f;
        for (int l = 0; l < 16; ++l) s += s_rt[((b0 + rr) * 16 + l) * 256 + u];
        xbuf[rr * P0 + u]        = f2bf(s * 0.0625f);
        xbuf[rr * P0 + 512 + u]  = 0;
        l1buf[rr * P1 + 256 + u] = 0;
    }
    __syncthreads();

    for (int t = 0; t < 512; ++t) {
        // stage A: audio tile -> xbuf[:,256:512] (bf16)
        {
            const float* ap = audio + ((size_t)(b0 + rr) * 512 + t) * 256 + uu;
            float4 a0 = *(const float4*)(ap);
            float4 a1 = *(const float4*)(ap + 4);
            u16t* xp = xbuf + rr * P0 + 256 + uu;
            xp[0] = f2bf(a0.x); xp[1] = f2bf(a0.y); xp[2] = f2bf(a0.z); xp[3] = f2bf(a0.w);
            xp[4] = f2bf(a1.x); xp[5] = f2bf(a1.y); xp[6] = f2bf(a1.z); xp[7] = f2bf(a1.w);
        }
        __syncthreads();

        // stage B: layer0 gates  z = [prev_out|a_t|h1] (16x768) @ W0cat (768x1024)
        gemm_tile(xbuf, P0, 24, w0, bias0, zbuf, lane, wave);
        __syncthreads();

        // stage C: layer0 nonlinearity (f32), c1 in registers
        {
            const float* zr = zbuf + rr * ZP;
#pragma unroll
            for (int j = 0; j < 8; ++j) {
                int u = uu + j;
                float ig = fsig(zr[u]);
                float fg = fsig(zr[256 + u]);
                float gg = ftanh(zr[512 + u]);
                float og = fsig(zr[768 + u]);
                c1[j] = fg * c1[j] + ig * gg;
                u16t hb = f2bf(og * ftanh(c1[j]));
                xbuf[rr * P0 + 512 + u] = hb;   // h1 for next step (layer0 K-tail)
                l1buf[rr * P1 + u]      = hb;   // layer1 input now
            }
        }
        __syncthreads();

        // stage D: layer1 gates  z = [h1n|h2prev] (16x512) @ W1cat (512x1024)
        gemm_tile(l1buf, P1, 16, w1, bias1, zbuf, lane, wave);
        __syncthreads();

        // stage E: layer1 nonlinearity, c2 in registers; h2 feedback
        {
            const float* zr = zbuf + rr * ZP;
#pragma unroll
            for (int j = 0; j < 8; ++j) {
                int u = uu + j;
                float ig = fsig(zr[u]);
                float fg = fsig(zr[256 + u]);
                float gg = ftanh(zr[512 + u]);
                float og = fsig(zr[768 + u]);
                c2[j] = fg * c2[j] + ig * gg;
                u16t hb = f2bf(og * ftanh(c2[j]));
                l1buf[rr * P1 + 256 + u] = hb;  // h2prev for next step
                xbuf[rr * P0 + u]        = hb;  // prev_out for next step
                h2st[rr * 256 + u]       = hb;  // staging for packed out write
            }
        }
        __syncthreads();

        // stage F: write h2 tile to global in head A-fragment packed layout
        {
            u32t* dst = outA + (size_t)(t * 16 + wg) * 2048;   // 8KB tile, dword units
#pragma unroll
            for (int i = 0; i < 4; ++i) {
                int d   = tid * 4 + i;
                int kb  = d >> 8;  int rem = d & 255;
                int ln  = rem >> 3; int v  = rem & 7;
                int row = ln & 15;
                int klo = ((v < 4) ? 2 * v : 16 + 2 * (v - 4)) + ((ln < 16) ? 0 : 8);
                int k   = kb * 32 + klo;
                u32t lo = h2st[row * 256 + k];
                u32t hi = h2st[row * 256 + k + 1];
                dst[d]  = lo | (hi << 16);
            }
        }
        __syncthreads();
    }
}

// ---------------- head GEMM: y = out @ W1^T + b1 ; accumulate BN stats ----------------
__global__ __launch_bounds__(256) void head_gemm(const uint4* __restrict__ outA,
                                                 const uint4* __restrict__ wh,
                                                 const float* __restrict__ b1v,
                                                 float* __restrict__ ybuf,
                                                 float* __restrict__ gsum, float* __restrict__ gsq)
{
    __shared__ float sred[128], sqred[128];
    const int tid = threadIdx.x, lane = tid & 31, wave = tid >> 5;
    if (tid < 128) { sred[tid] = 0.0f; sqred[tid] = 0.0f; }
    __syncthreads();

    const int m = blockIdx.x * 8 + wave;                  // M-tile 0..8191
    const uint4* abase = outA + (size_t)m * 512 + lane * 2;
    v8f acc[8];
#pragma unroll
    for (int nt = 0; nt < 8; ++nt) acc[nt] = (v8f){};
    for (int kb = 0; kb < 8; ++kb) {
        Frag a;
        a.q[0] = abase[kb * 64];
        a.q[1] = abase[kb * 64 + 1];
        const uint4* wp = wh + (size_t)(kb * 8) * 64 + lane * 2;
#pragma unroll
        for (int nt = 0; nt < 8; ++nt) {
            Frag b;
            b.q[0] = wp[nt * 64];
            b.q[1] = wp[nt * 64 + 1];
            acc[nt] = wmma_bf16(a.v, b.v, acc[nt]);
        }
    }
    const int mlo = (lane < 16) ? 0 : 8;
#pragma unroll
    for (int nt = 0; nt < 8; ++nt) {
        int n = nt * 16 + (lane & 15);
        float bb = b1v[n];
        float s1 = 0.0f, s2 = 0.0f;
#pragma unroll
        for (int v = 0; v < 8; ++v) {
            float y = acc[nt][v] + bb;
            ybuf[(size_t)(m * 16 + mlo + v) * 128 + n] = y;
            s1 += y; s2 += y * y;
        }
        atomicAdd(&sred[n], s1);
        atomicAdd(&sqred[n], s2);
    }
    __syncthreads();
    if (tid < 128) {
        atomicAdd(&gsum[tid], sred[tid]);
        atomicAdd(&gsq[tid],  sqred[tid]);
    }
}

__global__ void bn_finalize(const float* __restrict__ gsum, const float* __restrict__ gsq,
                            const float* __restrict__ gamma, const float* __restrict__ beta,
                            float* __restrict__ bna, float* __restrict__ bnb)
{
    int i = threadIdx.x;   // 128
    const float inv = 1.0f / 131072.0f;
    float mu  = gsum[i] * inv;
    float var = gsq[i] * inv - mu * mu;
    float a   = gamma[i] * rsqrtf(var + 1e-5f);
    bna[i] = a;
    bnb[i] = beta[i] - mu * a;
}

// ---------------- final: BN + ReLU + Linear(128->6), scatter to (B,T,6) ----------------
__global__ __launch_bounds__(256) void head_final(const float* __restrict__ ybuf,
                                                  const float* __restrict__ bna,
                                                  const float* __restrict__ bnb,
                                                  const float* __restrict__ W2,
                                                  const float* __restrict__ b2,
                                                  float* __restrict__ out)
{
    __shared__ float sw[6 * 128], sa[128], sb[128], sb2[8];
    const int tid = threadIdx.x;
    for (int i = tid; i < 768; i += 256) sw[i] = W2[i];
    if (tid < 128) { sa[tid] = bna[tid]; sb[tid] = bnb[tid]; }
    if (tid < 6)   sb2[tid] = b2[tid];
    __syncthreads();

    const int row = blockIdx.x * 256 + tid;     // t-major row' = t*256 + b
    const float* yp = ybuf + (size_t)row * 128;
    float acc[6] = {0, 0, 0, 0, 0, 0};
    for (int n = 0; n < 128; ++n) {
        float yn = fmaxf(yp[n] * sa[n] + sb[n], 0.0f);
#pragma unroll
        for (int j = 0; j < 6; ++j) acc[j] += yn * sw[j * 128 + n];
    }
    const int b = row & 255, t = row >> 8;
    float* op = out + (size_t)(b * 512 + t) * 6;
#pragma unroll
    for (int j = 0; j < 6; ++j) op[j] = acc[j] + sb2[j];
}

// ---------------- launch ----------------
extern "C" void kernel_launch(void* const* d_in, const int* in_sizes, int n_in,
                              void* d_out, int out_size, void* d_ws, size_t ws_size,
                              hipStream_t stream)
{
    const float* audio = (const float*)d_in[1];
    const float* s_rt  = (const float*)d_in[2];
    const float* Wih0  = (const float*)d_in[3];
    const float* Whh0  = (const float*)d_in[4];
    const float* bih0  = (const float*)d_in[5];
    const float* bhh0  = (const float*)d_in[6];
    const float* Wih1  = (const float*)d_in[7];
    const float* Whh1  = (const float*)d_in[8];
    const float* bih1  = (const float*)d_in[9];
    const float* bhh1  = (const float*)d_in[10];
    const float* W1    = (const float*)d_in[11];
    const float* b1v   = (const float*)d_in[12];
    const float* gamma = (const float*)d_in[13];
    const float* beta  = (const float*)d_in[14];
    const float* W2    = (const float*)d_in[15];
    const float* b2v   = (const float*)d_in[16];

    char* ws = (char*)d_ws;
    u32t*  w0p   = (u32t*)(ws + OFF_W0);
    u32t*  w1p   = (u32t*)(ws + OFF_W1);
    u32t*  whp   = (u32t*)(ws + OFF_WH);
    float* bias0 = (float*)(ws + OFF_B0);
    float* bias1 = (float*)(ws + OFF_B1);
    float* gsum  = (float*)(ws + OFF_SUM);
    float* gsq   = (float*)(ws + OFF_SQ);
    float* bna   = (float*)(ws + OFF_BNA);
    float* bnb   = (float*)(ws + OFF_BNB);
    u32t*  outA  = (u32t*)(ws + OFF_OUTA);
    float* ybuf  = (float*)(ws + OFF_Y);

    // 1) pack weights to bf16 fragments, fuse biases, zero BN accumulators
    {
        int total = 24 * 64 * 256 + 16 * 64 * 256 + 8 * 8 * 256 + 2048 + 256;
        int blocks = (total + 255) / 256;
        pack_kernel<<<blocks, 256, 0, stream>>>(Wih0, Whh0, bih0, bhh0,
                                                Wih1, Whh1, bih1, bhh1, W1,
                                                w0p, w1p, whp, bias0, bias1, gsum, gsq);
    }

    // 2) persistent LSTM decode: 16 batch tiles x 512 steps
    {
        size_t shmem = 41472 + (size_t)16 * ZP * 4 + 16 * 256 * 2;  // 115,712 B
        lstm_kernel<<<16, 512, shmem, stream>>>(audio, s_rt,
                                                (const uint4*)w0p, (const uint4*)w1p,
                                                bias0, bias1, outA);
    }

    // 3) head GEMM + BN stats
    head_gemm<<<1024, 256, 0, stream>>>((const uint4*)outA, (const uint4*)whp,
                                        b1v, ybuf, gsum, gsq);

    // 4) BN finalize
    bn_finalize<<<1, 128, 0, stream>>>(gsum, gsq, gamma, beta, bna, bnb);

    // 5) BN apply + ReLU + Linear(128->6), scatter to (B,T,6)
    head_final<<<512, 256, 0, stream>>>(ybuf, bna, bnb, W2, b2v, (float*)d_out);
}
